// PMHAttention_62019327754285
// MI455X (gfx1250) — matrix operand order repaired and checked by reading, not compiled, not run on hardware
//
#include <hip/hip_runtime.h>

// ---------------------------------------------------------------------------
// Problem dims (compile-time)
// ---------------------------------------------------------------------------
constexpr int B_ = 32, N_ = 256, E_ = 512, H_ = 8, ID_ = 8, D_ = 64;
constexpr int M_ = B_ * N_;          // 8192 rows for the projections
constexpr int K3E = 3 * E_;          // 1536

typedef _Float16 half8 __attribute__((ext_vector_type(8)));
typedef _Float16 v16h  __attribute__((ext_vector_type(16)));
typedef float    v8f   __attribute__((ext_vector_type(8)));

typedef unsigned int uint4v __attribute__((ext_vector_type(4)));
typedef int          int8v  __attribute__((ext_vector_type(8)));
typedef int          int4v  __attribute__((ext_vector_type(4)));

// ---------------------------------------------------------------------------
// WMMA wrapper: D = A(16x32 f16) * B(32x16 f16) + C(16x16 f32)
// ---------------------------------------------------------------------------
__device__ inline v8f wmma16(v16h a, v16h b, v8f c) {
    return __builtin_amdgcn_wmma_f32_16x16x32_f16(
        /*neg_a=*/false, a, /*neg_b=*/false, b,
        /*c_mod=*/(short)0, c, /*reuse_a=*/false, /*reuse_b=*/false);
}

// A-operand loader (16x32 tile, row-major source, contiguous-k A-layout):
//   lane l (lo=l&15, hi=l>>4) holds row M=lo,
//   K halves [k0+hi*16 .. +15] contiguous  (two 16B groups)
__device__ inline v16h load_a16(const _Float16* p, int stride, int row0, int k0,
                                int lo, int hi) {
    const _Float16* q = p + (size_t)(row0 + lo) * stride + k0 + hi * 16;
    union { half8 h[2]; v16h v; } u;
    u.h[0] = *(const half8*)q;
    u.h[1] = *(const half8*)(q + 8);
    return u.v;
}

// ---------------------------------------------------------------------------
// TDM: 1-D DMA of `nelem` f16 elements global -> LDS (one descriptor).
// D# built per CDNA5 ISA:
//   group0: count=1, lds_addr, global_addr[56:0], type=2
//   group1: data_size=1 (2B), tensor_dim0 = tile_dim0 = nelem,
//           tensor_dim1 = tile_dim1 = 1, dim0_stride = nelem
// Tracked on TENSORcnt. This toolchain's builtin is the 6-arg flavor.
// ---------------------------------------------------------------------------
__device__ inline void tdm_load_f16_1d(unsigned int lds_off, const void* gptr,
                                       unsigned int nelem) {
    unsigned long long ga = (unsigned long long)gptr;
    uint4v g0;
    g0[0] = 1u;                                        // count=1
    g0[1] = lds_off;                                   // lds_addr
    g0[2] = (unsigned int)(ga & 0xFFFFFFFFu);          // global_addr[31:0]
    g0[3] = (unsigned int)((ga >> 32) & 0x01FFFFFFu)   // global_addr[56:32]
            | (2u << 30);                              // type=2 ("image")
    int8v g1;
    g1[0] = (1 << 16);                                 // data_size=1 -> 2B
    g1[1] = (int)((nelem & 0xFFFFu) << 16);            // tensor_dim0 lo16
    g1[2] = (int)((nelem >> 16) & 0xFFFFu) | (1 << 16);// dim0 hi16 | tensor_dim1=1 lo
    g1[3] = (int)((nelem & 0xFFFFu) << 16);            // tensor_dim1 hi=0 | tile_dim0
    g1[4] = 1;                                         // tile_dim1=1, tile_dim2=0
    g1[5] = (int)nelem;                                // tensor_dim0_stride lo32
    g1[6] = 0;                                         // stride hi | dim1_stride lo
    g1[7] = 0;
    int4v gz4 = {0, 0, 0, 0};                          // groups 2/3 unused (2-D)
    int8v gz8 = {0, 0, 0, 0, 0, 0, 0, 0};
    __builtin_amdgcn_tensor_load_to_lds(g0, g1, gz4, gz4, gz8, /*cpol=*/0);
}

// ---------------------------------------------------------------------------
// Kernel 1: f32 -> f16 convert, 8 elems / thread
// ---------------------------------------------------------------------------
__global__ __launch_bounds__(256)
void pmha_cvt_f16(const float* __restrict__ src, _Float16* __restrict__ dst,
                  int n8) {
    int i = blockIdx.x * 256 + threadIdx.x;
    if (i >= n8) return;
    const float4* s = (const float4*)src + (size_t)i * 2;
    float4 x0 = s[0], x1 = s[1];
    half8 o;
    o[0] = (_Float16)x0.x; o[1] = (_Float16)x0.y;
    o[2] = (_Float16)x0.z; o[3] = (_Float16)x0.w;
    o[4] = (_Float16)x1.x; o[5] = (_Float16)x1.y;
    o[6] = (_Float16)x1.z; o[7] = (_Float16)x1.w;
    *(half8*)(dst + (size_t)i * 8) = o;
}

// ---------------------------------------------------------------------------
// Kernel 2: bias[b,h,n,m] = f16( U[b,n,m,:] . w_u[h,:] + b_u[h] )
// one thread per (b,n,m); reads U exactly once (32B), emits 8 f16
// ---------------------------------------------------------------------------
__global__ __launch_bounds__(256)
void pmha_bias(const float* __restrict__ U, const float* __restrict__ wu,
               const float* __restrict__ bu, _Float16* __restrict__ bias16) {
    __shared__ float wus[H_ * ID_];
    __shared__ float bus[H_];
    int tid = threadIdx.x;
    if (tid < H_ * ID_) wus[tid] = wu[tid];
    if (tid < H_) bus[tid] = bu[tid];
    __syncthreads();
    int idx = blockIdx.x * 256 + tid;            // < B*N*N = 2M
    int b = idx >> 16, n = (idx >> 8) & 255, m = idx & 255;
    const float4* up = (const float4*)(U + (size_t)idx * ID_);
    float4 x0 = up[0], x1 = up[1];
    float u[8] = {x0.x, x0.y, x0.z, x0.w, x1.x, x1.y, x1.z, x1.w};
    #pragma unroll
    for (int h = 0; h < H_; ++h) {
        float s = bus[h];
        #pragma unroll
        for (int i = 0; i < ID_; ++i) s += u[i] * wus[h * ID_ + i];
        bias16[(((size_t)(b * H_ + h) * N_) + n) * N_ + m] = (_Float16)s;
    }
}

// ---------------------------------------------------------------------------
// Kernel 3/5: WMMA GEMM  out[m,n] = A[m,:K] . W[n,:K] + bias[n]
//   block = 256 threads = 8 waves; wave computes 16(M) x 64(N); K chunk 32
//   mode 0: f32 store to out  (out-projection)
//   mode 1: scatter epilogue into Q (x0.125) / K [B,H,N,D] and Vt [B,H,D,N]
//           region = blockIdx.y (0=Q, 1=K, 2=V)
// ---------------------------------------------------------------------------
__global__ __launch_bounds__(256)
void pmha_gemm(const _Float16* __restrict__ A, const _Float16* __restrict__ W,
               const float* __restrict__ bias, int K, int mode,
               _Float16* __restrict__ qh, _Float16* __restrict__ kh,
               _Float16* __restrict__ vth, float* __restrict__ outF) {
    int tid = threadIdx.x, lane = tid & 31, wid = tid >> 5;
    int lo = lane & 15, hi = lane >> 4;
    int mbase = blockIdx.x * 16;
    int nbase = blockIdx.y * 512 + wid * 64;

    v8f acc[4];
    #pragma unroll
    for (int t = 0; t < 4; ++t) {
        float bv = bias[nbase + t * 16 + lo];
        #pragma unroll
        for (int r = 0; r < 8; ++r) acc[t][r] = bv;
    }

    for (int k0 = 0; k0 < K; k0 += 32) {
        v16h a = load_a16(A, K, mbase, k0, lo, hi);
        #pragma unroll
        for (int t = 0; t < 4; ++t) {
            // B operand: lane = column n, 16 contiguous K halves at hi*16
            v16h b = *(const v16h*)(W + (size_t)(nbase + t * 16 + lo) * K +
                                    k0 + hi * 16);
            acc[t] = wmma16(a, b, acc[t]);
        }
    }

    if (mode == 0) {
        #pragma unroll
        for (int t = 0; t < 4; ++t)
            #pragma unroll
            for (int r = 0; r < 8; ++r) {
                int m = mbase + r + 8 * hi;
                int n = nbase + t * 16 + lo;
                outF[(size_t)m * E_ + n] = acc[t][r];
            }
    } else {
        int region = blockIdx.y;                 // uniform: 0=Q 1=K 2=V
        #pragma unroll
        for (int t = 0; t < 4; ++t)
            #pragma unroll
            for (int r = 0; r < 8; ++r) {
                int n512 = wid * 64 + t * 16 + lo;   // col within E
                int h = n512 >> 6, d = n512 & 63;
                int m = mbase + r + 8 * hi;
                int b = m >> 8, nn = m & 255;
                float v = acc[t][r];
                if (region == 0) {               // Q, pre-scaled by D^-0.5
                    qh[(((size_t)(b * H_ + h) * N_) + nn) * D_ + d] =
                        (_Float16)(v * 0.125f);
                } else if (region == 1) {        // K row-major [N,D]
                    kh[(((size_t)(b * H_ + h) * N_) + nn) * D_ + d] =
                        (_Float16)v;
                } else {                         // V transposed [D,N]
                    vth[(((size_t)(b * H_ + h) * D_) + d) * N_ + nn] =
                        (_Float16)v;
                }
            }
    }
}

// ---------------------------------------------------------------------------
// Kernel 4: fused attention core, one workgroup per (b,h)
//   256 threads = 8 waves; wave handles 32 query rows (2 x 16-row tiles)
//   LDS: K tile 32KB + Vt tile 32KB + per-wave p scratch 64KB = 128KB
//   K / Vt tiles staged via Tensor Data Mover (TENSORcnt), no VGPR round-trip
// ---------------------------------------------------------------------------
__global__ __launch_bounds__(256)
void pmha_attn(const _Float16* __restrict__ qh, const _Float16* __restrict__ kh,
               const _Float16* __restrict__ vth,
               const _Float16* __restrict__ bias16, _Float16* __restrict__ oh) {
    __shared__ _Float16 ks[N_ * D_];       // K  [256][64]
    __shared__ _Float16 vs[D_ * N_];       // Vt [64][256]
    __shared__ _Float16 pb[8 * 16 * N_];   // per-wave probs [16][256]

    int bh = blockIdx.x;
    int b = bh >> 3, h = bh & 7;
    int tid = threadIdx.x, lane = tid & 31, wid = tid >> 5;
    int lo = lane & 15, hi = lane >> 4;

    // ---- TDM stage of K and Vt into LDS (one descriptor each, wave 0) ----
    if (wid == 0) {
        unsigned int ks_off = (unsigned int)(size_t)(void*)ks;
        unsigned int vs_off = (unsigned int)(size_t)(void*)vs;
        tdm_load_f16_1d(ks_off, kh + (size_t)bh * N_ * D_, N_ * D_);
        tdm_load_f16_1d(vs_off, vth + (size_t)bh * N_ * D_, N_ * D_);
        __builtin_amdgcn_s_wait_tensorcnt(0);
    }
    __syncthreads();

    const _Float16* qp = qh + (size_t)bh * N_ * D_;
    const _Float16* bp = bias16 + (size_t)bh * N_ * N_;
    _Float16* mypb = pb + wid * (16 * N_);

    for (int sub = 0; sub < 2; ++sub) {
        int q0 = (wid * 2 + sub) * 16;

        // Q A-operands for both K-chunks of D=64
        v16h aq0 = load_a16(qp, D_, q0, 0, lo, hi);
        v16h aq1 = load_a16(qp, D_, q0, 32, lo, hi);

        // ---- S = Qs.K^T + bias, kept in registers (C/D layout) ----
        float sreg[16][8];
        #pragma unroll
        for (int mt = 0; mt < 16; ++mt) {
            v8f c;
            #pragma unroll
            for (int r = 0; r < 8; ++r)
                c[r] = (float)bp[(size_t)(q0 + r + 8 * hi) * N_ + mt * 16 + lo];
            v16h b0 = *(const v16h*)(ks + (mt * 16 + lo) * D_ + 0 + hi * 16);
            v16h b1 = *(const v16h*)(ks + (mt * 16 + lo) * D_ + 32 + hi * 16);
            c = wmma16(aq0, b0, c);
            c = wmma16(aq1, b1, c);
            #pragma unroll
            for (int r = 0; r < 8; ++r) sreg[mt][r] = c[r];
        }

        // ---- softmax along m (cols live in lanes 0..15 of each half) ----
        float inv[8];
        #pragma unroll
        for (int r = 0; r < 8; ++r) {
            float mx = -3.0e38f;
            #pragma unroll
            for (int mt = 0; mt < 16; ++mt) mx = fmaxf(mx, sreg[mt][r]);
            #pragma unroll
            for (int d = 1; d < 16; d <<= 1)
                mx = fmaxf(mx, __shfl_xor(mx, d, 32));
            float sum = 0.f;
            #pragma unroll
            for (int mt = 0; mt < 16; ++mt) {
                float p = __expf(sreg[mt][r] - mx);
                sreg[mt][r] = p;
                sum += p;
            }
            #pragma unroll
            for (int d = 1; d < 16; d <<= 1) sum += __shfl_xor(sum, d, 32);
            inv[r] = 1.0f / sum;
        }

        // ---- C-layout -> A-layout bounce through per-wave LDS ----
        #pragma unroll
        for (int mt = 0; mt < 16; ++mt)
            #pragma unroll
            for (int r = 0; r < 8; ++r)
                mypb[(r + 8 * hi) * N_ + mt * 16 + lo] =
                    (_Float16)sreg[mt][r];

        // ---- O = P.V  (K dim = m = 256, 8 chunks of 32) ----
        v8f oacc[4];
        #pragma unroll
        for (int t = 0; t < 4; ++t)
            #pragma unroll
            for (int r = 0; r < 8; ++r) oacc[t][r] = 0.f;

        #pragma unroll
        for (int mc = 0; mc < 8; ++mc) {
            v16h a = load_a16(mypb, N_, 0, mc * 32, lo, hi);
            #pragma unroll
            for (int dt = 0; dt < 4; ++dt) {
                v16h bm = *(const v16h*)(vs + (dt * 16 + lo) * N_ +
                                         mc * 32 + hi * 16);
                oacc[dt] = wmma16(a, bm, oacc[dt]);
            }
        }

        // ---- normalize + store merged-head f16 output [B,N,E] ----
        #pragma unroll
        for (int dt = 0; dt < 4; ++dt)
            #pragma unroll
            for (int r = 0; r < 8; ++r) {
                int row = q0 + r + 8 * hi;
                int col = h * D_ + dt * 16 + lo;
                oh[((size_t)(b * N_ + row)) * E_ + col] =
                    (_Float16)(oacc[dt][r] * inv[r]);
            }
    }
}

// ---------------------------------------------------------------------------
// Launcher
// ---------------------------------------------------------------------------
extern "C" void kernel_launch(void* const* d_in, const int* in_sizes, int n_in,
                              void* d_out, int out_size, void* d_ws,
                              size_t ws_size, hipStream_t stream) {
    const float* x     = (const float*)d_in[0];   // [B,N,E]
    const float* U     = (const float*)d_in[1];   // [B,N,N,ID]
    const float* wqkv  = (const float*)d_in[2];   // [3E,E]
    const float* bqkv  = (const float*)d_in[3];   // [3E]
    const float* wout  = (const float*)d_in[4];   // [E,E]
    const float* bout  = (const float*)d_in[5];   // [E]
    const float* wu    = (const float*)d_in[6];   // [H,ID]
    const float* bu    = (const float*)d_in[7];   // [H]
    float* out = (float*)d_out;                   // [B,N,E]

    char* ws = (char*)d_ws;
    size_t o = 0;
    _Float16* xh    = (_Float16*)(ws + o); o += (size_t)M_ * E_ * 2;    // 8 MB
    _Float16* wqkvh = (_Float16*)(ws + o); o += (size_t)K3E * E_ * 2;   // 1.5 MB
    _Float16* wouth = (_Float16*)(ws + o); o += (size_t)E_ * E_ * 2;    // 0.5 MB
    _Float16* qhh   = (_Float16*)(ws + o); o += (size_t)M_ * E_ * 2;    // 8 MB
    _Float16* khh   = (_Float16*)(ws + o); o += (size_t)M_ * E_ * 2;    // 8 MB
    _Float16* vth   = (_Float16*)(ws + o); o += (size_t)M_ * E_ * 2;    // 8 MB
    _Float16* b16   = (_Float16*)(ws + o); o += (size_t)B_ * H_ * N_ * N_ * 2; // 32 MB
    _Float16* ohh   = (_Float16*)(ws + o); o += (size_t)M_ * E_ * 2;    // 8 MB
    (void)o; (void)ws_size; (void)in_sizes; (void)n_in; (void)out_size;

    // 1) f32 -> f16 conversions
    pmha_cvt_f16<<<(M_ * E_ / 8 + 255) / 256, 256, 0, stream>>>(x, xh, M_ * E_ / 8);
    pmha_cvt_f16<<<(K3E * E_ / 8 + 255) / 256, 256, 0, stream>>>(wqkv, wqkvh, K3E * E_ / 8);
    pmha_cvt_f16<<<(E_ * E_ / 8 + 255) / 256, 256, 0, stream>>>(wout, wouth, E_ * E_ / 8);

    // 2) per-head additive bias from U (U read exactly once)
    pmha_bias<<<B_ * N_ * N_ / 256, 256, 0, stream>>>(U, wu, bu, b16);

    // 3) QKV projection -> Q(scaled)/K [B,H,N,D], Vt [B,H,D,N]
    pmha_gemm<<<dim3(M_ / 16, 3), 256, 0, stream>>>(
        xh, wqkvh, bqkv, E_, /*mode=*/1, qhh, khh, vth, nullptr);

    // 4) fused attention core, one WG per (b,h)
    pmha_attn<<<B_ * H_, 256, 0, stream>>>(qhh, khh, vth, b16, ohh);

    // 5) output projection -> f32 d_out
    pmha_gemm<<<dim3(M_ / 16, 1), 256, 0, stream>>>(
        ohh, wouth, bout, E_, /*mode=*/0, nullptr, nullptr, nullptr, out);
}
